// CausalSelfAttention_54528904790511
// MI455X (gfx1250) — matrix operand order repaired and checked by reading, not compiled, not run on hardware
//
#include <hip/hip_runtime.h>
#include <hip/hip_bf16.h>

typedef __attribute__((ext_vector_type(16))) __bf16 v16bf;
typedef __attribute__((ext_vector_type(8)))  __bf16 v8bf;
typedef __attribute__((ext_vector_type(4)))  __bf16 v4bf;
typedef __attribute__((ext_vector_type(8)))  float  v8f;

#define B_DIM  2
#define T_DIM  2048
#define C_DIM  1024
#define H_DIM  16
#define HS_DIM 64

__device__ __forceinline__ v8f wmma_bf16(v16bf a, v16bf b, v8f c) {
  return __builtin_amdgcn_wmma_f32_16x16x32_bf16(false, a, false, b, (short)0, c,
                                                 false, false);
}
__device__ __forceinline__ v16bf cat16(v8bf lo, v8bf hi) {
  return __builtin_shufflevector(lo, hi, 0, 1, 2, 3, 4, 5, 6, 7,
                                 8, 9, 10, 11, 12, 13, 14, 15);
}
__device__ __forceinline__ v8bf ld8(const __bf16* p) { return *(const v8bf*)p; }

// ---------------------------------------------------------------------------
// Kernel 1: qkv = x @ W_qkv + b_qkv, RoPE(q,k), scatter:
//   q,k -> [B*H, T, HS] bf16 (rope applied),  v -> [B*H, HS, T] bf16 (transposed)
// M=4096, K=1024, N=3072. 128 thr = 4 waves, tile 64x64, K-step 64.
// LDS: As[m][k] row-major, Bs[n][k] TRANSPOSED so B-fragments are contiguous.
// ---------------------------------------------------------------------------
__global__ __launch_bounds__(128)
void qkv_rope_kernel(const float* __restrict__ x, const float* __restrict__ W,
                     const float* __restrict__ bias,
                     __bf16* __restrict__ qo, __bf16* __restrict__ ko,
                     __bf16* __restrict__ vt)
{
  __shared__ __bf16 As[64][72];   // [m][k]  row stride 144B (16B-aligned)
  __shared__ __bf16 Bs[64][72];   // [n][k]  transposed weight tile
  __shared__ float  Cs[64][65];

  const int Ntiles = (3 * C_DIM) / 64;            // 48
  const int tile_n = blockIdx.x % Ntiles;
  const int tile_m = blockIdx.x / Ntiles;
  const int m0 = tile_m * 64, n0 = tile_n * 64;
  const int ldb = 3 * C_DIM;

  const int tid  = threadIdx.x;
  const int lane = tid & 31, wv = tid >> 5;
  const int l15  = lane & 15, lh = lane >> 4;
  const int wm   = (wv >> 1) * 32, wn = (wv & 1) * 32;
  const int base = lh * 16;

  v8f c[2][2];
#pragma unroll
  for (int mi = 0; mi < 2; ++mi)
#pragma unroll
    for (int ni = 0; ni < 2; ++ni)
#pragma unroll
      for (int e = 0; e < 8; ++e) c[mi][ni][e] = 0.f;

  for (int k0 = 0; k0 < C_DIM; k0 += 64) {
#pragma unroll
    for (int i = 0; i < 8; ++i) {                 // x tile 64x64 f32 -> bf16
      int idx = tid + i * 128;                    // 1024 float4 chunks
      int r = idx >> 4, c4 = (idx & 15) * 4;
      const float4 v = *(const float4*)(x + (size_t)(m0 + r) * C_DIM + k0 + c4);
      v4bf t = {(__bf16)v.x, (__bf16)v.y, (__bf16)v.z, (__bf16)v.w};
      *(v4bf*)&As[r][c4] = t;
    }
#pragma unroll
    for (int i = 0; i < 8; ++i) {                 // W tile 64x64 -> Bs[n][k]
      int idx = tid + i * 128;
      int r = idx >> 4, c4 = (idx & 15) * 4;      // r = k row, c4 = n offset
      const float4 v = *(const float4*)(W + (size_t)(k0 + r) * ldb + n0 + c4);
      Bs[c4 + 0][r] = (__bf16)v.x; Bs[c4 + 1][r] = (__bf16)v.y;
      Bs[c4 + 2][r] = (__bf16)v.z; Bs[c4 + 3][r] = (__bf16)v.w;
    }
    __syncthreads();

    v16bf a[2][2], b[2][2];
#pragma unroll
    for (int mi = 0; mi < 2; ++mi) {
      const __bf16* ap = &As[wm + mi * 16 + l15][0];
#pragma unroll
      for (int kc = 0; kc < 2; ++kc)
        a[mi][kc] = cat16(ld8(ap + kc * 32 + base),
                          ld8(ap + kc * 32 + base + 8));
    }
#pragma unroll
    for (int ni = 0; ni < 2; ++ni) {
      const __bf16* bp = &Bs[wn + ni * 16 + l15][0];
#pragma unroll
      for (int kc = 0; kc < 2; ++kc)
        b[ni][kc] = cat16(ld8(bp + kc * 32 + lh * 16),
                          ld8(bp + kc * 32 + lh * 16 + 8));
    }
#pragma unroll
    for (int kc = 0; kc < 2; ++kc)
#pragma unroll
      for (int mi = 0; mi < 2; ++mi)
#pragma unroll
        for (int ni = 0; ni < 2; ++ni)
          c[mi][ni] = wmma_bf16(a[mi][kc], b[ni][kc], c[mi][ni]);
    __syncthreads();
  }

  // bias + stage f32 tile (C layout: vgpr j -> row j + 8*lanehalf)
#pragma unroll
  for (int mi = 0; mi < 2; ++mi)
#pragma unroll
    for (int ni = 0; ni < 2; ++ni) {
      float bv = bias[n0 + wn + ni * 16 + l15];
#pragma unroll
      for (int j = 0; j < 8; ++j)
        Cs[wm + mi * 16 + j + lh * 8][wn + ni * 16 + l15] = c[mi][ni][j] + bv;
    }
  __syncthreads();

  const int b_idx = m0 / T_DIM;
  const int t0    = m0 % T_DIM;
  const int h     = n0 / (3 * HS_DIM);
  const int part  = (n0 % (3 * HS_DIM)) / HS_DIM; // 0=q 1=k 2=v

  if (part < 2) {
    __bf16* dst = (part == 0 ? qo : ko) +
                  (size_t)(b_idx * H_DIM + h) * T_DIM * HS_DIM;
    for (int idx = tid; idx < 64 * 32; idx += 128) {
      int tl = idx >> 5, p = idx & 31;
      float x0 = Cs[tl][2 * p], x1 = Cs[tl][2 * p + 1];
      float theta = __powf(10000.f, -(float)p * (1.f / 32.f));
      float ang = (float)(t0 + tl) * theta;
      float sn, cs;
      sincosf(ang, &sn, &cs);
      size_t o = (size_t)(t0 + tl) * HS_DIM + 2 * p;
      dst[o]     = (__bf16)(x0 * cs - x1 * sn);
      dst[o + 1] = (__bf16)(x0 * sn + x1 * cs);
    }
  } else {
    __bf16* dst = vt + (size_t)(b_idx * H_DIM + h) * HS_DIM * T_DIM;
    for (int idx = tid; idx < 64 * 64; idx += 128) {
      int d = idx >> 6, tl = idx & 63;
      dst[(size_t)d * T_DIM + t0 + tl] = (__bf16)Cs[tl][d];
    }
  }
}

// ---------------------------------------------------------------------------
// Kernel 2: flash attention. 4 independent waves/block; each wave owns 16
// query rows of one (b,h). Keys 32 at a time: S = QK^T (4 WMMA, K=64),
// online softmax, P@V (4 WMMA, K=32) over 64 head dims. All fragment loads
// are 16B vector loads.
// ---------------------------------------------------------------------------
__global__ __launch_bounds__(128)
void attn_kernel(const __bf16* __restrict__ qb, const __bf16* __restrict__ kb,
                 const __bf16* __restrict__ vt, __bf16* __restrict__ yb)
{
  __shared__ __bf16 plds[4][16][40];              // per-wave P bounce, 80B rows

  const int tid  = threadIdx.x;
  const int lane = tid & 31, wv = tid >> 5;
  const int l15  = lane & 15, lh = lane >> 4;
  const int qtile = blockIdx.x & 31;              // T/64 = 32
  const int bh    = blockIdx.x >> 5;              // 0..31
  const int qw    = qtile * 64 + wv * 16;

  const __bf16* qh = qb + (size_t)bh * T_DIM * HS_DIM;
  const __bf16* kh = kb + (size_t)bh * T_DIM * HS_DIM;
  const __bf16* vh = vt + (size_t)bh * HS_DIM * T_DIM;

  const int c0 = lh * 16;
  v16bf qf[2];                                    // Q A-fragments K=0..31,32..63
  {
    const __bf16* qp = qh + (size_t)(qw + l15) * HS_DIM;
#pragma unroll
    for (int f = 0; f < 2; ++f)
      qf[f] = cat16(ld8(qp + f * 32 + c0), ld8(qp + f * 32 + c0 + 8));
  }

  v8f acc[4];
#pragma unroll
  for (int nt = 0; nt < 4; ++nt)
#pragma unroll
    for (int e = 0; e < 8; ++e) acc[nt][e] = 0.f;
  float mrun[8], lrun[8];
#pragma unroll
  for (int j = 0; j < 8; ++j) { mrun[j] = -3.0e38f; lrun[j] = 0.f; }

  const int qlast = qw + 15;
  const float scale = 0.125f;                     // 1/sqrt(64)

  for (int k0 = 0; k0 <= qlast; k0 += 32) {
    v8f s[2];
#pragma unroll
    for (int sub = 0; sub < 2; ++sub) {           // 16 keys each
      const __bf16* kp = kh + (size_t)(k0 + sub * 16 + l15) * HS_DIM + lh * 16;
      v16bf kf0 = cat16(ld8(kp), ld8(kp + 8));
      v16bf kf1 = cat16(ld8(kp + 32), ld8(kp + 40));
      v8f cz;
#pragma unroll
      for (int e = 0; e < 8; ++e) cz[e] = 0.f;
      cz = wmma_bf16(qf[0], kf0, cz);
      cz = wmma_bf16(qf[1], kf1, cz);
      s[sub] = cz;
    }

#pragma unroll
    for (int j = 0; j < 8; ++j) {                 // online softmax per row
      const int q = qw + j + lh * 8;
      float v0 = s[0][j] * scale; if (k0 + l15 > q)      v0 = -1.0e30f;
      float v1 = s[1][j] * scale; if (k0 + 16 + l15 > q) v1 = -1.0e30f;
      float mx = fmaxf(v0, v1);
      mx = fmaxf(mx, __shfl_xor(mx, 1, 32));
      mx = fmaxf(mx, __shfl_xor(mx, 2, 32));
      mx = fmaxf(mx, __shfl_xor(mx, 4, 32));
      mx = fmaxf(mx, __shfl_xor(mx, 8, 32));
      const float nm   = fmaxf(mrun[j], mx);
      const float corr = __expf(mrun[j] - nm);
      const float p0 = __expf(v0 - nm);
      const float p1 = __expf(v1 - nm);
      float sm = p0 + p1;
      sm += __shfl_xor(sm, 1, 32);
      sm += __shfl_xor(sm, 2, 32);
      sm += __shfl_xor(sm, 4, 32);
      sm += __shfl_xor(sm, 8, 32);
      lrun[j] = lrun[j] * corr + sm;
      mrun[j] = nm;
#pragma unroll
      for (int nt = 0; nt < 4; ++nt) acc[nt][j] *= corr;
      const int mrow = j + lh * 8;
      plds[wv][mrow][l15]      = (__bf16)p0;      // C-layout -> LDS (transpose)
      plds[wv][mrow][16 + l15] = (__bf16)p1;
    }
    asm volatile("s_wait_dscnt 0" ::: "memory");  // LDS RAW (same wave)

    const __bf16* pp = &plds[wv][l15][0];
    v16bf pf = cat16(ld8(pp + c0), ld8(pp + c0 + 8));
#pragma unroll
    for (int nt = 0; nt < 4; ++nt) {              // P @ V (K=32, contiguous v^T)
      const __bf16* vp = vh + (size_t)(nt * 16 + l15) * T_DIM + k0 + lh * 16;
      v16bf vf = cat16(ld8(vp), ld8(vp + 8));
      acc[nt] = wmma_bf16(pf, vf, acc[nt]);
    }
  }

  const int b = bh >> 4, h = bh & 15;
#pragma unroll
  for (int j = 0; j < 8; ++j) {
    const float inv = 1.f / lrun[j];
    const int q = qw + j + lh * 8;
    __bf16* yp = yb + ((size_t)(b * T_DIM + q)) * C_DIM + h * HS_DIM;
#pragma unroll
    for (int nt = 0; nt < 4; ++nt)
      yp[nt * 16 + l15] = (__bf16)(acc[nt][j] * inv);
  }
}

// ---------------------------------------------------------------------------
// Kernel 3: out = y_att @ W_proj + b_proj.  M=4096, K=1024, N=1024, Kstep 64.
// ---------------------------------------------------------------------------
__global__ __launch_bounds__(128)
void proj_kernel(const __bf16* __restrict__ A, const float* __restrict__ W,
                 const float* __restrict__ bias, float* __restrict__ out)
{
  __shared__ __bf16 As[64][72];   // [m][k]
  __shared__ __bf16 Bs[64][72];   // [n][k] transposed

  const int Ntiles = C_DIM / 64;                  // 16
  const int tile_n = blockIdx.x % Ntiles;
  const int tile_m = blockIdx.x / Ntiles;
  const int m0 = tile_m * 64, n0 = tile_n * 64;

  const int tid  = threadIdx.x;
  const int lane = tid & 31, wv = tid >> 5;
  const int l15  = lane & 15, lh = lane >> 4;
  const int wm   = (wv >> 1) * 32, wn = (wv & 1) * 32;
  const int base = lh * 16;

  v8f c[2][2];
#pragma unroll
  for (int mi = 0; mi < 2; ++mi)
#pragma unroll
    for (int ni = 0; ni < 2; ++ni)
#pragma unroll
      for (int e = 0; e < 8; ++e) c[mi][ni][e] = 0.f;

  for (int k0 = 0; k0 < C_DIM; k0 += 64) {
#pragma unroll
    for (int i = 0; i < 4; ++i) {                 // A tile 64x64 bf16, 16B loads
      int idx = tid + i * 128;                    // 512 chunks of 8 bf16
      int r = idx >> 3, c8 = (idx & 7) * 8;
      *(v8bf*)&As[r][c8] = ld8(A + (size_t)(m0 + r) * C_DIM + k0 + c8);
    }
#pragma unroll
    for (int i = 0; i < 8; ++i) {                 // W tile 64x64 -> Bs[n][k]
      int idx = tid + i * 128;
      int r = idx >> 4, c4 = (idx & 15) * 4;
      const float4 v = *(const float4*)(W + (size_t)(k0 + r) * C_DIM + n0 + c4);
      Bs[c4 + 0][r] = (__bf16)v.x; Bs[c4 + 1][r] = (__bf16)v.y;
      Bs[c4 + 2][r] = (__bf16)v.z; Bs[c4 + 3][r] = (__bf16)v.w;
    }
    __syncthreads();

    v16bf a[2][2], b[2][2];
#pragma unroll
    for (int mi = 0; mi < 2; ++mi) {
      const __bf16* ap = &As[wm + mi * 16 + l15][0];
#pragma unroll
      for (int kc = 0; kc < 2; ++kc)
        a[mi][kc] = cat16(ld8(ap + kc * 32 + base),
                          ld8(ap + kc * 32 + base + 8));
    }
#pragma unroll
    for (int ni = 0; ni < 2; ++ni) {
      const __bf16* bp = &Bs[wn + ni * 16 + l15][0];
#pragma unroll
      for (int kc = 0; kc < 2; ++kc)
        b[ni][kc] = cat16(ld8(bp + kc * 32 + lh * 16),
                          ld8(bp + kc * 32 + lh * 16 + 8));
    }
#pragma unroll
    for (int kc = 0; kc < 2; ++kc)
#pragma unroll
      for (int mi = 0; mi < 2; ++mi)
#pragma unroll
        for (int ni = 0; ni < 2; ++ni)
          c[mi][ni] = wmma_bf16(a[mi][kc], b[ni][kc], c[mi][ni]);
    __syncthreads();
  }

#pragma unroll
  for (int mi = 0; mi < 2; ++mi)
#pragma unroll
    for (int ni = 0; ni < 2; ++ni) {
      const int col = n0 + wn + ni * 16 + l15;
      const float bv = bias[col];
#pragma unroll
      for (int j = 0; j < 8; ++j) {
        const int row = m0 + wm + mi * 16 + j + lh * 8;
        out[(size_t)row * C_DIM + col] = c[mi][ni][j] + bv;
      }
    }
}

// ---------------------------------------------------------------------------
extern "C" void kernel_launch(void* const* d_in, const int* in_sizes, int n_in,
                              void* d_out, int out_size, void* d_ws, size_t ws_size,
                              hipStream_t stream) {
  const float* x      = (const float*)d_in[0];
  const float* W_qkv  = (const float*)d_in[1];
  const float* b_qkv  = (const float*)d_in[2];
  const float* W_proj = (const float*)d_in[3];
  const float* b_proj = (const float*)d_in[4];
  float* out = (float*)d_out;

  const size_t QE = (size_t)B_DIM * H_DIM * T_DIM * HS_DIM; // 4,194,304
  __bf16* qbuf  = (__bf16*)d_ws;
  __bf16* kbuf  = qbuf + QE;
  __bf16* vtbuf = kbuf + QE;
  __bf16* ybuf  = vtbuf + QE;                               // 32 MB total

  qkv_rope_kernel<<<dim3(64 * 48), dim3(128), 0, stream>>>(
      x, W_qkv, b_qkv, qbuf, kbuf, vtbuf);
  attn_kernel<<<dim3(B_DIM * H_DIM * (T_DIM / 64)), dim3(128), 0, stream>>>(
      qbuf, kbuf, vtbuf, ybuf);
  proj_kernel<<<dim3(64 * 16), dim3(128), 0, stream>>>(
      ybuf, W_proj, b_proj, out);
}
